// DiffJPEG_62895501083057
// MI455X (gfx1250) — matrix operand order repaired and checked by reading, not compiled, not run on hardware
//
#include <hip/hip_runtime.h>
#include <hip/hip_bf16.h>

typedef __attribute__((ext_vector_type(16))) _Float16 v16h;
typedef __attribute__((ext_vector_type(8)))  _Float16 v8h;
typedef __attribute__((ext_vector_type(2)))  _Float16 h2;
typedef __attribute__((ext_vector_type(8)))  float    v8f;

#define RS2  0.70710678118654752f   // 1/sqrt(2)
#define FACT 0.4f                   // quality factor for q=80
#define PI16 0.19634954084936207f   // pi/16

// One workgroup = one 32x32 pixel region. 6 waves: w0..w3 -> four 16x16 Y
// tiles, w4 -> Cb, w5 -> Cr (pooled 16x16). diag(C8,C8) block-diagonal basis
// makes one 16x16 WMMA = four 8x8 DCTs. Chain is arranged so data is always
// the A operand (contiguous ds_load_b128 per lane) and the cosine matrix is
// always the B operand (register constants, no LDS):
//   F1: U  = X  * Md^T        F2: D^T = U^T * Md^T
//   I1: P  = dd * Md          I2: R^T = P^T * Md
// Storing the transpose of a C/D-layout result is a contiguous ds_store_b128
// and yields exactly the row-major matrix the next stage's A operand reads.
// Quant/dequant scales are precomputed into LDS (no divisions in hot path).
__global__ __launch_bounds__(192) void diffjpeg_kernel(
    const float* __restrict__ img, const float* __restrict__ ytab,
    const float* __restrict__ ctab, float* __restrict__ out)
{
    __shared__ _Float16 Yl [32*32];      // Y - 128, f16, row-major
    __shared__ _Float16 Cbl[16*16];      // pooled Cb - 128
    __shared__ _Float16 Crl[16*16];      // pooled Cr - 128
    __shared__ float    SQ[128];         // 0.25*au*av/(tab*FACT)  [0..63]=Y, [64..127]=C
    __shared__ float    SD[128];         // tab*FACT*au*av
    __shared__ _Float16 Tw [6][16*16];   // per-wave matmul-stage scratch
    __shared__ float    Yr [32*32];      // reconstructed Y
    __shared__ float    Cbr[16*16];      // reconstructed Cb
    __shared__ float    Crr[16*16];      // reconstructed Cr

    const int tid = threadIdx.x;
    const int bb  = blockIdx.x >> 8;         // batch index (32)
    const int tt  = blockIdx.x & 255;        // 16x16 grid of 32x32 regions
    const int R0  = (tt >> 4) << 5;
    const int C0  = (tt & 15) << 5;
    const size_t plane = 512u * 512u;
    const size_t ibase = (size_t)bb * 3u * plane;

    if (tid < 128) {
        int e = tid & 63;                      // (u,v) in 8x8
        float tabv = (tid < 64) ? ytab[e] : ctab[e];
        float tf   = tabv * FACT;
        float au   = ((e >> 3) == 0) ? RS2 : 1.f;
        float av   = ((e & 7) == 0) ? RS2 : 1.f;
        SQ[tid] = 0.25f * au * av / tf;        // one fdiv, setup only
        SD[tid] = tf * au * av;
    }

    // ---------- Phase 1: load 32x32 RGB, convert, pool chroma ----------
    for (int q = tid; q < 256; q += 192) {
        int qr = q >> 4, qc = q & 15;
        int lr = qr << 1, lc = qc << 1;
        int gr = R0 + lr, gc = C0 + lc;
        const float* p0 = img + ibase;
        float2 rr0 = *(const float2*)(p0 + 0 * plane + (size_t)gr * 512 + gc);
        float2 rr1 = *(const float2*)(p0 + 0 * plane + (size_t)(gr + 1) * 512 + gc);
        float2 gg0 = *(const float2*)(p0 + 1 * plane + (size_t)gr * 512 + gc);
        float2 gg1 = *(const float2*)(p0 + 1 * plane + (size_t)(gr + 1) * 512 + gc);
        float2 bb0 = *(const float2*)(p0 + 2 * plane + (size_t)gr * 512 + gc);
        float2 bb1 = *(const float2*)(p0 + 2 * plane + (size_t)(gr + 1) * 512 + gc);
        float cbs = 0.f, crs = 0.f;
        auto yval = [&](float R, float G, float B) {
            R *= 255.f; G *= 255.f; B *= 255.f;
            cbs += -0.168736f * R - 0.331264f * G + 0.5f * B;
            crs += 0.5f * R - 0.418688f * G - 0.081312f * B;
            return 0.299f * R + 0.587f * G + 0.114f * B - 128.f;
        };
        h2 t;
        t.x = (_Float16)yval(rr0.x, gg0.x, bb0.x);
        t.y = (_Float16)yval(rr0.y, gg0.y, bb0.y);
        *(h2*)&Yl[lr * 32 + lc] = t;
        t.x = (_Float16)yval(rr1.x, gg1.x, bb1.x);
        t.y = (_Float16)yval(rr1.y, gg1.y, bb1.y);
        *(h2*)&Yl[(lr + 1) * 32 + lc] = t;
        // cb/cr computed without +128, so pooled value is already (ch - 128)
        Cbl[qr * 16 + qc] = (_Float16)(0.25f * cbs);
        Crl[qr * 16 + qc] = (_Float16)(0.25f * crs);
    }
    __syncthreads();

    // ---------- Phase 2: per-wave WMMA DCT -> quant -> dequant -> IDCT ----------
    const int w    = tid >> 5;
    const int lane = tid & 31;
    const int m    = lane & 15;      // A row / B,C,D column index
    const int hi   = lane >> 4;      // lane-half selects K chunk
    const int n7   = m & 7;
    const int nblk = m >> 3;

    const _Float16* src; int sstr; int toff; float* rec; int rstr;
    if (w < 4)      { src = Yl  + ((w >> 1) * 16) * 32 + (w & 1) * 16; sstr = 32;
                      toff = 0;  rec = Yr + ((w >> 1) * 16) * 32 + (w & 1) * 16; rstr = 32; }
    else if (w == 4){ src = Cbl; sstr = 16; toff = 64; rec = Cbr; rstr = 16; }
    else            { src = Crl; sstr = 16; toff = 64; rec = Crr; rstr = 16; }
    _Float16* tw = Tw[w];

    // Register-resident B constants (ISA 16-bit B layout: lane = column n,
    // elements 0..7 K=8*hi+j, elements 8..15 K=16+8*hi+(j-8), zero padding here).
    v16h bMdT, bMd;
    #pragma unroll
    for (int j = 0; j < 16; ++j) {
        int k2 = (j < 8) ? 8 * hi + j : 16 + 8 * hi + (j - 8);
        bool ok = (k2 < 16) && ((k2 >> 3) == nblk);
        float aT = __cosf((float)(2 * (k2 & 7) + 1) * (float)n7 * PI16); // Md^T[k][n]=C[n&7][k&7]
        float aM = __cosf((float)(2 * n7 + 1) * (float)(k2 & 7) * PI16); // Md  [k][n]=C[k&7][n&7]
        bMdT[j] = (_Float16)(ok ? aT : 0.f);
        bMd [j] = (_Float16)(ok ? aM : 0.f);
    }

    // A operand: row m, 8 contiguous halfs at K=8*hi (one ds_load_b128), K>=16 pad 0.
    auto loadA = [&](const _Float16* p, int stride) {
        v8h d = *(const v8h*)(p + m * stride + 8 * hi);
        v16h a;
        #pragma unroll
        for (int j = 0; j < 8; ++j) { a[j] = d[j]; a[j + 8] = (_Float16)0.f; }
        return a;
    };
    // Store transpose of C/D result: lds[N*16+M], M=g+8*hi contiguous -> one b128.
    auto storeT = [&](_Float16* p, v8f z) {
        v8h s;
        #pragma unroll
        for (int g = 0; g < 8; ++g) s[g] = (_Float16)z[g];
        *(v8h*)(p + m * 16 + 8 * hi) = s;
    };

    v16h a;
    v8f  acc, zz = (v8f){0.f,0.f,0.f,0.f,0.f,0.f,0.f,0.f};

    // F1: U = X * Md^T
    a   = loadA(src, sstr);
    acc = __builtin_amdgcn_wmma_f32_16x16x32_f16(false, a, false, bMdT, (short)0, acc = zz, false, false);
    storeT(tw, acc);

    // F2: D^T = U^T * Md^T
    a   = loadA(tw, 16);
    acc = __builtin_amdgcn_wmma_f32_16x16x32_f16(false, a, false, bMdT, (short)0, acc = zz, false, false);

    // Quantize / diff_round / dequantize.  Lane holds D[u][v] with u = m,
    // v = g + 8*hi  =>  u%8 = n7, v%8 = g; scale rows are contiguous in LDS.
    {
        const float* sq = SQ + toff + n7 * 8;
        const float* sd = SD + toff + n7 * 8;
        const float4 q0 = *(const float4*)(sq);
        const float4 q1 = *(const float4*)(sq + 4);
        const float4 d0 = *(const float4*)(sd);
        const float4 d1 = *(const float4*)(sd + 4);
        v8h s;
        #pragma unroll
        for (int g = 0; g < 8; ++g) {
            float sqv = (g < 4) ? (&q0.x)[g] : (&q1.x)[g - 4];
            float sdv = (g < 4) ? (&d0.x)[g] : (&d1.x)[g - 4];
            float d  = acc[g] * sqv;
            float rn = rintf(d);
            float e  = d - rn;
            float dd = (rn + e * e * e) * sdv;            // diff_round + dequant
            s[g] = (_Float16)dd;
        }
        *(v8h*)(tw + m * 16 + 8 * hi) = s;                // dd row-major [u][v]
    }

    // I1: P = dd * Md
    a   = loadA(tw, 16);
    acc = __builtin_amdgcn_wmma_f32_16x16x32_f16(false, a, false, bMd, (short)0, acc = zz, false, false);
    storeT(tw, acc);

    // I2: R^T = P^T * Md ; transpose-store => R row-major spatial, pixel = 0.25*R + 128
    a   = loadA(tw, 16);
    acc = __builtin_amdgcn_wmma_f32_16x16x32_f16(false, a, false, bMd, (short)0, acc = zz, false, false);
    {
        float* p = rec + m * rstr + 8 * hi;
        float4 f0 = { 0.25f*acc[0]+128.f, 0.25f*acc[1]+128.f, 0.25f*acc[2]+128.f, 0.25f*acc[3]+128.f };
        float4 f1 = { 0.25f*acc[4]+128.f, 0.25f*acc[5]+128.f, 0.25f*acc[6]+128.f, 0.25f*acc[7]+128.f };
        *(float4*)(p)     = f0;
        *(float4*)(p + 4) = f1;
    }
    __syncthreads();

    // ---------- Phase 3: upsample chroma, YCbCr -> RGB, clip, store ----------
    for (int q = tid; q < 256; q += 192) {
        int qr = q >> 4, qc = q & 15;
        int lr = qr << 1, lc = qc << 1;
        int gr = R0 + lr, gc = C0 + lc;
        float cb = Cbr[qr * 16 + qc] - 128.f;
        float cr = Crr[qr * 16 + qc] - 128.f;
        float rA = 1.402f * cr;
        float gA = -0.344136f * cb - 0.714136f * cr;
        float bA = 1.772f * cb;
        float2 yA = *(const float2*)&Yr[lr * 32 + lc];
        float2 yB = *(const float2*)&Yr[(lr + 1) * 32 + lc];
        auto fin = [](float y, float add) {
            float v = y + add;
            return fminf(fmaxf(v, 0.f), 255.f) * (1.f / 255.f);
        };
        float* o = out + ibase;
        float2 v2;
        v2.x = fin(yA.x, rA); v2.y = fin(yA.y, rA);
        *(float2*)(o + 0 * plane + (size_t)gr * 512 + gc) = v2;
        v2.x = fin(yB.x, rA); v2.y = fin(yB.y, rA);
        *(float2*)(o + 0 * plane + (size_t)(gr + 1) * 512 + gc) = v2;
        v2.x = fin(yA.x, gA); v2.y = fin(yA.y, gA);
        *(float2*)(o + 1 * plane + (size_t)gr * 512 + gc) = v2;
        v2.x = fin(yB.x, gA); v2.y = fin(yB.y, gA);
        *(float2*)(o + 1 * plane + (size_t)(gr + 1) * 512 + gc) = v2;
        v2.x = fin(yA.x, bA); v2.y = fin(yA.y, bA);
        *(float2*)(o + 2 * plane + (size_t)gr * 512 + gc) = v2;
        v2.x = fin(yB.x, bA); v2.y = fin(yB.y, bA);
        *(float2*)(o + 2 * plane + (size_t)(gr + 1) * 512 + gc) = v2;
    }
}

extern "C" void kernel_launch(void* const* d_in, const int* in_sizes, int n_in,
                              void* d_out, int out_size, void* d_ws, size_t ws_size,
                              hipStream_t stream) {
    const float* img = (const float*)d_in[0];   // (32,3,512,512) f32
    const float* yt  = (const float*)d_in[1];   // (8,8) f32
    const float* ct  = (const float*)d_in[2];   // (8,8) f32
    float* out = (float*)d_out;                 // (32,3,512,512) f32
    dim3 grid(32 * 256);                        // 32 batches x 256 regions of 32x32
    dim3 block(192);                            // 6 waves of 32
    diffjpeg_kernel<<<grid, block, 0, stream>>>(img, yt, ct, out);
}
